// HierarchicalMambaBlock_21895743275186
// MI455X (gfx1250) — compile-verified, matrix operand order checked by reading, not run on hardware
//
#include <hip/hip_runtime.h>
#include <hip/hip_bf16.h>
#include <math.h>

// ---------------------------------------------------------------------------
// HierarchicalMambaBlock for MI455X (gfx1250, wave32).
// Big GEMMs run on v_wmma_f32_16x16x32_f16 (f16 inputs, f32 accumulate),
// 32x64 wave tile (2 A-frags x 4 B-frags -> 8 WMMA per K-step).
// ---------------------------------------------------------------------------

typedef _Float16 half_t;
typedef __attribute__((ext_vector_type(16))) _Float16 v16h;
typedef __attribute__((ext_vector_type(8)))  float    v8f;

#define BB   4
#define TT   4096
#define DIMC 1024
#define DI   2048
#define BT   (BB * TT)        // 16384 token rows

__device__ __forceinline__ float sigmoidf_(float x) { return 1.0f / (1.0f + __expf(-x)); }
__device__ __forceinline__ float siluf_(float x)    { return x * sigmoidf_(x); }
__device__ __forceinline__ float softplusf_(float x) {
    if (x > 20.0f)  return x;
    if (x < -20.0f) return __expf(x);
    return log1pf(__expf(x));
}

// ------------------------------- casts -------------------------------------
__global__ void cast_f32_to_f16(const float* __restrict__ in, half_t* __restrict__ out, int n) {
    int i = blockIdx.x * blockDim.x + threadIdx.x;
    if (i < n) out[i] = (half_t)in[i];
}

// W is (K x N) row major; write Wt (N x K) f16 so GEMM B-fragments are contiguous.
__global__ void transpose_cast_f16(const float* __restrict__ W, half_t* __restrict__ Wt,
                                   int K, int N) {
    int i = blockIdx.x * blockDim.x + threadIdx.x;
    if (i < K * N) {
        int k = i / N, n = i % N;
        Wt[(size_t)n * K + k] = (half_t)W[i];
    }
}

// ------------------------------- WMMA GEMM ---------------------------------
// C(M,N) = A(M,K) * Bt(N,K)^T.
// Wave tile 32x64 (2 M-subtiles x 4 N-subtiles, 8 v8f accumulators).
// Block: 8 waves = 4 (M) x 2 (N) -> block tile 128x128.
// EPI 0: n <  N/2 -> outA[m*(N/2)+n]      = f16(acc)            (x_in)
//        n >= N/2 -> outB[m*(N/2)+n-N/2]  = f16(silu(acc))      (silu(gate))
// EPI 1: outA[m*N+n] = f16(silu(acc))
// EPI 2: outA[m*N+n] = f16( auxF[m*N+n] * sigmoid(acc) * (float)auxH[m*N+n] )
// EPI 3: outF[m*N+n] = acc + auxF[m*N+n]                        (+ residual)
template <int EPI>
__global__ __launch_bounds__(256)
void gemm_wmma_f16(const half_t* __restrict__ A, const half_t* __restrict__ Bt,
                   int M, int N, int K,
                   half_t* __restrict__ outA, half_t* __restrict__ outB,
                   float* __restrict__ outF,
                   const float* __restrict__ auxF, const half_t* __restrict__ auxH) {
    const int lane = threadIdx.x & 31;
    const int wave = threadIdx.x >> 5;
    const int wm   = wave & 3;       // 4 waves along M
    const int wn   = wave >> 2;      // 2 waves along N
    const int lr   = lane & 15;      // row-in-tile (A) / col-in-tile (B,C)
    const int hb   = lane >> 4;      // half-wave selects K sub-range
    const int m0   = blockIdx.y * 128 + wm * 32;
    const int n0   = blockIdx.x * 128 + wn * 64;

    v8f acc00 = {}, acc01 = {}, acc02 = {}, acc03 = {};
    v8f acc10 = {}, acc11 = {}, acc12 = {}, acc13 = {};

    const half_t* arow0 = A  + (size_t)(m0 + lr) * K + hb * 16;
    const half_t* arow1 = arow0 + (size_t)16 * K;
    const half_t* brow  = Bt + (size_t)(n0 + lr) * K + hb * 16;
    const size_t bstep  = (size_t)16 * K;

    for (int kb = 0; kb < K; kb += 32) {
        v16h a0 = *(const v16h*)(arow0 + kb);
        v16h a1 = *(const v16h*)(arow1 + kb);
        v16h b0 = *(const v16h*)(brow + kb);
        v16h b1 = *(const v16h*)(brow + bstep     + kb);
        v16h b2 = *(const v16h*)(brow + bstep * 2 + kb);
        v16h b3 = *(const v16h*)(brow + bstep * 3 + kb);
        if (kb + 32 < K) {                    // prefetch next K slab, near cache
            __builtin_prefetch(arow0 + kb + 32, 0, 3);
            __builtin_prefetch(arow1 + kb + 32, 0, 3);
            __builtin_prefetch(brow  + kb + 32, 0, 3);
        }
        acc00 = __builtin_amdgcn_wmma_f32_16x16x32_f16(false, a0, false, b0, (short)0, acc00, false, false);
        acc01 = __builtin_amdgcn_wmma_f32_16x16x32_f16(false, a0, false, b1, (short)0, acc01, false, false);
        acc02 = __builtin_amdgcn_wmma_f32_16x16x32_f16(false, a0, false, b2, (short)0, acc02, false, false);
        acc03 = __builtin_amdgcn_wmma_f32_16x16x32_f16(false, a0, false, b3, (short)0, acc03, false, false);
        acc10 = __builtin_amdgcn_wmma_f32_16x16x32_f16(false, a1, false, b0, (short)0, acc10, false, false);
        acc11 = __builtin_amdgcn_wmma_f32_16x16x32_f16(false, a1, false, b1, (short)0, acc11, false, false);
        acc12 = __builtin_amdgcn_wmma_f32_16x16x32_f16(false, a1, false, b2, (short)0, acc12, false, false);
        acc13 = __builtin_amdgcn_wmma_f32_16x16x32_f16(false, a1, false, b3, (short)0, acc13, false, false);
    }

    // C layout: VGPR r, lanes 0-15 -> M=r, N=lane ; lanes 16-31 -> M=8+r.
    const int half_n = N >> 1;
#pragma unroll
    for (int i = 0; i < 2; ++i) {
        v8f* accs[4] = { i == 0 ? &acc00 : &acc10, i == 0 ? &acc01 : &acc11,
                         i == 0 ? &acc02 : &acc12, i == 0 ? &acc03 : &acc13 };
#pragma unroll
        for (int r = 0; r < 8; ++r) {
            const int m = m0 + i * 16 + hb * 8 + r;
            const size_t rowoff = (size_t)m * N;
#pragma unroll
            for (int j = 0; j < 4; ++j) {
                const int n = n0 + j * 16 + lr;
                const float v = (*accs[j])[r];
                if (EPI == 0) {
                    if (n < half_n) outA[(size_t)m * half_n + n] = (half_t)v;
                    else            outB[(size_t)m * half_n + (n - half_n)] = (half_t)siluf_(v);
                } else if (EPI == 1) {
                    outA[rowoff + n] = (half_t)siluf_(v);
                } else if (EPI == 2) {
                    outA[rowoff + n] = (half_t)(auxF[rowoff + n] * sigmoidf_(v) * (float)auxH[rowoff + n]);
                } else {
                    outF[rowoff + n] = v + auxF[rowoff + n];
                }
            }
        }
    }
}

// ------------------------------ downsample ---------------------------------
// mean-pool over time with stride st, per batch (input at full T resolution).
__global__ void downsample_k(const half_t* __restrict__ in, half_t* __restrict__ out,
                             int Ts, int st) {
    int i = blockIdx.x * blockDim.x + threadIdx.x;
    int n = BB * Ts * DI;
    if (i >= n) return;
    int c = i % DI;
    int t = (i / DI) % Ts;
    int b = i / (DI * Ts);
    float s = 0.0f;
    size_t base = ((size_t)b * TT + (size_t)t * st) * DI + c;
    for (int j = 0; j < st; ++j) s += (float)in[base + (size_t)j * DI];
    out[i] = (half_t)(s / (float)st);
}

// --------------------------- causal dwconv + silu ---------------------------
__global__ void conv_silu_k(const half_t* __restrict__ in, half_t* __restrict__ out,
                            const float* __restrict__ w, const float* __restrict__ bias,
                            int Ts) {
    int i = blockIdx.x * blockDim.x + threadIdx.x;
    int n = BB * Ts * DI;
    if (i >= n) return;
    int c = i % DI;
    int t = (i / DI) % Ts;
    int b = i / (DI * Ts);
    float acc = bias[c];
    const float* wc = w + (size_t)c * 4;
#pragma unroll
    for (int k = 0; k < 4; ++k) {
        int tt = t - 3 + k;
        if (tt >= 0) acc += wc[k] * (float)in[((size_t)b * Ts + tt) * DI + c];
    }
    out[i] = (half_t)siluf_(acc);
}

// --------------------- fused xproj(N=16) + dt + ssm output ------------------
// One block per token row: Bs[k] = sum_c xc[c]*xw[c,k] (k<16), then
// y[c] = xc[c]*sigmoid(softplus(Bs . dtw[:,c] + dtb[c])) + D[c]*x_s[c].
__global__ __launch_bounds__(256)
void ssm_row_k(const half_t* __restrict__ xc, const half_t* __restrict__ xs,
               const float* __restrict__ xw,   // (DI x 32), first 16 cols used
               const float* __restrict__ dtw,  // (16 x DI)
               const float* __restrict__ dtb, const float* __restrict__ Dp,
               half_t* __restrict__ ybuf) {
    const int row = blockIdx.x;
    const int tid = threadIdx.x;
    const half_t* xr = xc + (size_t)row * DI;
    __shared__ float red[16][256];
    __shared__ float Bs[16];

    float p[16];
#pragma unroll
    for (int k = 0; k < 16; ++k) p[k] = 0.0f;
    for (int c = tid; c < DI; c += 256) {
        float xv = (float)xr[c];
        const float* wrow = xw + (size_t)c * 32;
#pragma unroll
        for (int k = 0; k < 16; ++k) p[k] += xv * wrow[k];
    }
#pragma unroll
    for (int k = 0; k < 16; ++k) red[k][tid] = p[k];
    __syncthreads();
    for (int off = 128; off > 0; off >>= 1) {
        if (tid < off) {
#pragma unroll
            for (int k = 0; k < 16; ++k) red[k][tid] += red[k][tid + off];
        }
        __syncthreads();
    }
    if (tid < 16) Bs[tid] = red[tid][0];
    __syncthreads();

    float bsr[16];
#pragma unroll
    for (int k = 0; k < 16; ++k) bsr[k] = Bs[k];

    for (int c = tid; c < DI; c += 256) {
        float acc = dtb[c];
#pragma unroll
        for (int k = 0; k < 16; ++k) acc += bsr[k] * dtw[(size_t)k * DI + c];
        float dt = softplusf_(acc);
        float y  = (float)xr[c] * sigmoidf_(dt) + Dp[c] * (float)xs[(size_t)row * DI + c];
        ybuf[(size_t)row * DI + c] = (half_t)y;
    }
}

// ------------------- linear upsample + fused/ctx accumulate -----------------
__global__ void upsample_acc_k(const half_t* __restrict__ y,
                               float* __restrict__ fused, float* __restrict__ ctx,
                               const float* __restrict__ sw, int s, int Ts, int writeMode) {
    int i = blockIdx.x * blockDim.x + threadIdx.x;
    int n = BB * TT * DI;
    if (i >= n) return;
    // softmax over 3 scale weights (tiny, recomputed per thread)
    float e0 = __expf(sw[0]), e1 = __expf(sw[1]), e2 = __expf(sw[2]);
    float wsc = __expf(sw[s]) / (e0 + e1 + e2);

    int c = i % DI;
    int t = (i / DI) % TT;
    int b = i / (DI * TT);
    float scale = (float)Ts / (float)TT;
    float coord = fminf(fmaxf(((float)t + 0.5f) * scale - 0.5f, 0.0f), (float)(Ts - 1));
    int lo = (int)floorf(coord);
    int hi = min(lo + 1, Ts - 1);
    float w = coord - (float)lo;
    float v = (float)y[((size_t)b * Ts + lo) * DI + c] * (1.0f - w)
            + (float)y[((size_t)b * Ts + hi) * DI + c] * w;
    if (writeMode) {
        fused[i] = wsc * v;
        ctx[i]   = v * (1.0f / 3.0f);
    } else {
        fused[i] += wsc * v;
        ctx[i]   += v * (1.0f / 3.0f);
    }
}

// ------------------------------- LayerNorm ----------------------------------
__global__ __launch_bounds__(256)
void layernorm_k(const float* __restrict__ y, const float* __restrict__ g,
                 const float* __restrict__ b, float* __restrict__ out) {
    const int row = blockIdx.x;
    const int tid = threadIdx.x;
    const float* yr = y + (size_t)row * DIMC;
    float vals[4];
    float s = 0.0f, s2 = 0.0f;
#pragma unroll
    for (int j = 0; j < 4; ++j) {
        float v = yr[tid + j * 256];
        vals[j] = v;
        s += v; s2 += v * v;
    }
    __shared__ float rs[256], rq[256];
    rs[tid] = s; rq[tid] = s2;
    __syncthreads();
    for (int off = 128; off > 0; off >>= 1) {
        if (tid < off) { rs[tid] += rs[tid + off]; rq[tid] += rq[tid + off]; }
        __syncthreads();
    }
    float mu  = rs[0] * (1.0f / DIMC);
    float var = rq[0] * (1.0f / DIMC) - mu * mu;
    float inv = rsqrtf(var + 1e-5f);
#pragma unroll
    for (int j = 0; j < 4; ++j) {
        int col = tid + j * 256;
        out[(size_t)row * DIMC + col] = (vals[j] - mu) * inv * g[col] + b[col];
    }
}

// ------------------------------- launcher -----------------------------------
extern "C" void kernel_launch(void* const* d_in, const int* in_sizes, int n_in,
                              void* d_out, int out_size, void* d_ws, size_t ws_size,
                              hipStream_t stream) {
    (void)in_sizes; (void)n_in; (void)out_size; (void)ws_size;
    const float* x          = (const float*)d_in[0];
    const float* in_proj_w  = (const float*)d_in[1];
    const float* conv_w     = (const float*)d_in[2];
    const float* conv_b     = (const float*)d_in[3];
    const float* xproj_w    = (const float*)d_in[4];
    const float* dtproj_w   = (const float*)d_in[5];
    const float* dtproj_b   = (const float*)d_in[6];
    const float* D_param    = (const float*)d_in[7];
    const float* scale_w    = (const float*)d_in[8];
    const float* cg_w1      = (const float*)d_in[9];
    const float* cg_w2      = (const float*)d_in[10];
    const float* out_proj_w = (const float*)d_in[11];
    const float* ln_g       = (const float*)d_in[12];
    const float* ln_b       = (const float*)d_in[13];
    float* out = (float*)d_out;

    char* ws = (char*)d_ws;
    size_t off = 0;
    auto alloc = [&](size_t bytes) { size_t r = off; off += (bytes + 255) & ~(size_t)255; return r; };

    half_t* XH       = (half_t*)(ws + alloc((size_t)BT * DIMC * 2));     // x in f16
    half_t* WT_IN    = (half_t*)(ws + alloc((size_t)DIMC * 2 * DI * 2)); // in_proj^T
    half_t* WT_CG1   = (half_t*)(ws + alloc((size_t)DI * DIMC * 2));
    half_t* WT_CG2   = (half_t*)(ws + alloc((size_t)DIMC * DI * 2));
    half_t* WT_OUT   = (half_t*)(ws + alloc((size_t)DI * DIMC * 2));
    half_t* XIN_H    = (half_t*)(ws + alloc((size_t)BT * DI * 2));       // x_in f16
    half_t* GSILU_H  = (half_t*)(ws + alloc((size_t)BT * DI * 2));       // silu(gate)
    half_t* DS_H     = (half_t*)(ws + alloc((size_t)(BT / 2) * DI * 2)); // downsampled
    half_t* XC_H     = (half_t*)(ws + alloc((size_t)BT * DI * 2));       // conv+silu out
    half_t* YBUF_H   = (half_t*)(ws + alloc((size_t)BT * DI * 2));       // ssm out
    float*  FUSED_F  = (float*)(ws + alloc((size_t)BT * DI * 4));
    float*  CTX_F    = (float*)(ws + alloc((size_t)BT * DI * 4));
    // aliases (regions free after the scale loop):
    half_t* CTX_H    = XC_H;                 // f16 ctx for cg1 GEMM
    half_t* H1_H     = XH;                   // silu(ctx @ cg_w1)
    half_t* AOUT_H   = XIN_H;                // fused * cg * silu(gate)
    float*  Y_F      = FUSED_F;              // pre-LN output

    const int TPB = 256;
    auto blocks = [&](long long n) { return (unsigned)((n + TPB - 1) / TPB); };

    // 1) casts / weight transposes
    cast_f32_to_f16<<<blocks((long long)BT * DIMC), TPB, 0, stream>>>(x, XH, BT * DIMC);
    transpose_cast_f16<<<blocks((long long)DIMC * 2 * DI), TPB, 0, stream>>>(in_proj_w,  WT_IN,  DIMC, 2 * DI);
    transpose_cast_f16<<<blocks((long long)DI * DIMC),     TPB, 0, stream>>>(cg_w1,      WT_CG1, DI,   DIMC);
    transpose_cast_f16<<<blocks((long long)DIMC * DI),     TPB, 0, stream>>>(cg_w2,      WT_CG2, DIMC, DI);
    transpose_cast_f16<<<blocks((long long)DI * DIMC),     TPB, 0, stream>>>(out_proj_w, WT_OUT, DI,   DIMC);

    // 2) xz = x @ in_proj_w ; split -> x_in (f16), silu(gate) (f16)
    {
        dim3 g((2 * DI) / 128, BT / 128);
        gemm_wmma_f16<0><<<g, TPB, 0, stream>>>(XH, WT_IN, BT, 2 * DI, DIMC,
                                                XIN_H, GSILU_H, nullptr, nullptr, nullptr);
    }

    // 3) per-scale SSM
    for (int s = 0; s < 3; ++s) {
        const int st = 1 << s;
        const int Ts = TT >> s;
        const long long nEl = (long long)BB * Ts * DI;
        const half_t* xsrc = XIN_H;
        if (s > 0) {
            downsample_k<<<blocks(nEl), TPB, 0, stream>>>(XIN_H, DS_H, Ts, st);
            xsrc = DS_H;
        }
        conv_silu_k<<<blocks(nEl), TPB, 0, stream>>>(xsrc, XC_H,
                                                     conv_w + (size_t)s * DI * 4,
                                                     conv_b + (size_t)s * DI, Ts);
        ssm_row_k<<<BB * Ts, TPB, 0, stream>>>(XC_H, xsrc,
                                               xproj_w  + (size_t)s * DI * 32,
                                               dtproj_w + (size_t)s * 16 * DI,
                                               dtproj_b + (size_t)s * DI,
                                               D_param  + (size_t)s * DI, YBUF_H);
        upsample_acc_k<<<blocks((long long)BB * TT * DI), TPB, 0, stream>>>(
            YBUF_H, FUSED_F, CTX_F, scale_w, s, Ts, (s == 0) ? 1 : 0);
    }

    // 4) ctx -> f16, then cg pipeline on WMMA
    cast_f32_to_f16<<<blocks((long long)BT * DI), TPB, 0, stream>>>(CTX_F, CTX_H, BT * DI);
    {
        dim3 g(DIMC / 128, BT / 128);  // h1 = silu(ctx @ cg_w1)
        gemm_wmma_f16<1><<<g, TPB, 0, stream>>>(CTX_H, WT_CG1, BT, DIMC, DI,
                                                H1_H, nullptr, nullptr, nullptr, nullptr);
    }
    {
        dim3 g(DI / 128, BT / 128);    // aout = fused * sigmoid(h1 @ cg_w2) * silu(gate)
        gemm_wmma_f16<2><<<g, TPB, 0, stream>>>(H1_H, WT_CG2, BT, DI, DIMC,
                                                AOUT_H, nullptr, nullptr, FUSED_F, GSILU_H);
    }
    {
        dim3 g(DIMC / 128, BT / 128);  // y = aout @ out_proj_w + x
        gemm_wmma_f16<3><<<g, TPB, 0, stream>>>(AOUT_H, WT_OUT, BT, DIMC, DI,
                                                nullptr, nullptr, Y_F, x, nullptr);
    }

    // 5) LayerNorm over DIM
    layernorm_k<<<BT, TPB, 0, stream>>>(Y_F, ln_g, ln_b, out);
}
